// AdaptiveSpikingAttention_73023033967139
// MI455X (gfx1250) — 16-bit and 8-bit matrix operand order repaired and checked by reading, compiled, not run on hardware
//
#include <hip/hip_runtime.h>


#define B_    4
#define S_    512
#define E_    512
#define H_    8
#define HD_   64
#define TMAX_ 20
#define KPK_  (TMAX_ * HD_)
#define NTOK_ (B_ * S_)

typedef __attribute__((ext_vector_type(16))) _Float16 v16h;
typedef __attribute__((ext_vector_type(8)))  _Float16 v8h;
typedef __attribute__((ext_vector_type(8)))  float    v8f;
typedef __attribute__((ext_vector_type(8)))  int      v8i;

__device__ __forceinline__ v16h asa_afrag(const _Float16* row, int k0, int abase) {
  v8h lo = *(const v8h*)(row + k0 + 2 * abase);
  v8h hh = *(const v8h*)(row + k0 + 2 * abase + 8);
  v16h a;
#pragma unroll
  for (int u = 0; u < 8; ++u) { a[u] = lo[u]; a[u + 8] = hh[u]; }
  return a;
}

__global__ void asa_cvt_f16(const float* __restrict__ src,
                            _Float16* __restrict__ dst, int n) {
  int i = blockIdx.x * 256 + threadIdx.x;
  if (i < n) dst[i] = (_Float16)src[i];
}

__global__ void asa_transpose_w_f16(const float* __restrict__ W,
                                    _Float16* __restrict__ WT) {
  int idx = blockIdx.x * 256 + threadIdx.x;
  if (idx >= E_ * E_) return;
  int n = idx / E_;
  int k = idx - n * E_;
  WT[n * E_ + k] = (_Float16)W[k * E_ + n];
}

__global__ void asa_gate(const float* __restrict__ x,
                         const float* __restrict__ g1, const float* __restrict__ gb1,
                         const float* __restrict__ g2, const float* __restrict__ gb2,
                         const float* __restrict__ g3, const float* __restrict__ gb3,
                         const float* __restrict__ c1, const float* __restrict__ cb1,
                         const float* __restrict__ c2, const float* __restrict__ cb2,
                         int* __restrict__ Ti) {
  __shared__ float h1[64];
  __shared__ float h2[32];
  __shared__ float ch[32];
  int n = blockIdx.x;
  int j = threadIdx.x;
  const float* xr = x + (size_t)n * E_;

  float acc = gb1[j];
  for (int i = 0; i < E_; ++i) acc += xr[i] * g1[i * 64 + j];
  h1[j] = fmaxf(acc, 0.0f);
  if (j < 32) {
    float a2 = cb1[j];
    for (int i = 0; i < E_; ++i) a2 += xr[i] * c1[i * 32 + j];
    ch[j] = fmaxf(a2, 0.0f);
  }
  __syncthreads();
  if (j < 32) {
    float a = gb2[j];
    for (int i = 0; i < 64; ++i) a += h1[i] * g2[i * 32 + j];
    h2[j] = fmaxf(a, 0.0f);
  }
  __syncthreads();
  if (j == 0) {
    float sg = gb3[0], sc = cb2[0];
    for (int i = 0; i < 32; ++i) { sg += h2[i] * g3[i]; sc += ch[i] * c2[i]; }
    float gate = 1.0f / (1.0f + __expf(-sg));
    float comp = 1.0f / (1.0f + __expf(-sc));
    float comb = 0.7f * gate + 0.3f * comp;
    int t = (int)ceilf(comb * (float)TMAX_);
    t = t < 1 ? 1 : (t > TMAX_ ? TMAX_ : t);
    Ti[n] = t;
  }
}

__global__ void asa_gemm_f16(const _Float16* __restrict__ A,
                             const _Float16* __restrict__ BT,
                             float* __restrict__ C,
                             const float* __restrict__ bias,
                             int M, int N, int K) {
  int tile = blockIdx.x;
  int ntn  = N >> 5;
  int mi   = tile / ntn;
  int ni   = tile - mi * ntn;
  int lane = threadIdx.x & 31;
  int r    = lane & 15;
  int hi   = lane >> 4;
  int abase = hi * 8;
  int bbase = hi * 16;

  const _Float16* ar0 = A  + (size_t)(mi * 32 + r) * K;
  const _Float16* ar1 = ar0 + (size_t)16 * K;
  const _Float16* br0 = BT + (size_t)(ni * 32 + r) * K;
  const _Float16* br1 = br0 + (size_t)16 * K;

  v8f acc[2][2] = {};
  for (int k0 = 0; k0 < K; k0 += 32) {
    v16h a0 = asa_afrag(ar0, k0, abase);
    v16h a1 = asa_afrag(ar1, k0, abase);
    v16h b0 = *(const v16h*)(br0 + k0 + bbase);
    v16h b1 = *(const v16h*)(br1 + k0 + bbase);
    acc[0][0] = __builtin_amdgcn_wmma_f32_16x16x32_f16(false, a0, false, b0,
                                                       (short)0, acc[0][0], false, false);
    acc[0][1] = __builtin_amdgcn_wmma_f32_16x16x32_f16(false, a0, false, b1,
                                                       (short)0, acc[0][1], false, false);
    acc[1][0] = __builtin_amdgcn_wmma_f32_16x16x32_f16(false, a1, false, b0,
                                                       (short)0, acc[1][0], false, false);
    acc[1][1] = __builtin_amdgcn_wmma_f32_16x16x32_f16(false, a1, false, b1,
                                                       (short)0, acc[1][1], false, false);
  }
#pragma unroll
  for (int tj = 0; tj < 2; ++tj) {
    int nn = ni * 32 + tj * 16 + r;
    float bv = bias ? bias[nn] : 0.0f;
#pragma unroll
    for (int ti = 0; ti < 2; ++ti) {
#pragma unroll
      for (int v = 0; v < 8; ++v) {
        int mm = mi * 32 + ti * 16 + v + 8 * hi;
        C[(size_t)mm * N + nn] = acc[ti][tj][v] + bv;
      }
    }
  }
}

__global__ void asa_lif_pack(const float* __restrict__ q,
                             const float* __restrict__ k,
                             const float* __restrict__ v,
                             const int* __restrict__ Ti,
                             signed char* __restrict__ qp,
                             signed char* __restrict__ kp,
                             _Float16* __restrict__ vmT,
                             const float* __restrict__ alpha_p,
                             const float* __restrict__ beta_p) {
  int idx = blockIdx.x * 256 + threadIdx.x;
  if (idx >= NTOK_ * E_) return;
  int e = idx & (E_ - 1);
  int n = idx >> 9;
  int b = n >> 9;
  int s = n & (S_ - 1);
  int h = e >> 6;
  int d = e & (HD_ - 1);
  int Tn = Ti[n];
  float alpha = alpha_p[0], beta = beta_p[0];

  float xq = q[idx], xk = k[idx], xv = v[idx];
  float vq = 0.f, iq = 0.f, vk = 0.f, ik = 0.f, vv = 0.f, iv = 0.f;
  float vsum = 0.f;
  size_t rowoff = ((size_t)((b * H_ + h) * S_ + s)) * KPK_;
  signed char* qrow = qp + rowoff;
  signed char* krow = kp + rowoff;

#pragma unroll
  for (int t = 0; t < TMAX_; ++t) {
    float m = (t < Tn) ? 1.0f : 0.0f;
    iq = alpha * iq + xq; vq = beta * vq + iq;
    float sq = (vq >= 1.0f) ? 1.0f : 0.0f; vq *= (1.0f - sq);
    ik = alpha * ik + xk; vk = beta * vk + ik;
    float sk = (vk >= 1.0f) ? 1.0f : 0.0f; vk *= (1.0f - sk);
    iv = alpha * iv + xv; vv = beta * vv + iv;
    float sv = (vv >= 1.0f) ? 1.0f : 0.0f; vv *= (1.0f - sv);
    qrow[t * HD_ + d] = (signed char)(sq * m);
    krow[t * HD_ + d] = (signed char)(sk * m);
    vsum += sv * m;
  }
  vmT[((size_t)(b * H_ + h) * HD_ + d) * S_ + s] =
      (_Float16)(vsum * (1.0f / (float)TMAX_));
}

__global__ void asa_scores_iu8(const signed char* __restrict__ qp,
                               const signed char* __restrict__ kp,
                               float* __restrict__ scores) {
  __shared__ int skp[64 * KPK_ / 4];
  int blk = blockIdx.x;
  int jb = blk & 7;  blk >>= 3;
  int ib = blk & 7;  blk >>= 3;
  int bh = blk;
  int tid  = threadIdx.x;
  int wave = tid >> 5;
  int lane = tid & 31;

  const signed char* ksrc = kp + (size_t)(bh * S_ + jb * 64) * KPK_;
  for (int c = tid; c < (64 * KPK_) / 16; c += 128) {
    unsigned ldsoff = (unsigned)(size_t)(void*)(&skp[c * 4]);
    const void* g = (const void*)(ksrc + (size_t)c * 16);
    asm volatile("global_load_async_to_lds_b128 %0, %1, off"
                 :: "v"(ldsoff), "v"(g) : "memory");
  }
  asm volatile("s_wait_asynccnt 0" ::: "memory");
  __syncthreads();

  int r  = lane & 15;
  int hi = lane >> 4;
  int asub = hi * 16;
  int bsub = hi * 16;
  const signed char* qrow =
      qp + (size_t)(bh * S_ + ib * 64 + wave * 16 + r) * KPK_;

  v8i acc[4] = {};
  for (int kk = 0; kk < TMAX_; ++kk) {
    int k0 = kk * 64;
    v8i a;
    const int* pa;
    pa = (const int*)(qrow + k0 + asub);       a[0] = pa[0]; a[1] = pa[1]; a[2] = pa[2]; a[3] = pa[3];
    pa = (const int*)(qrow + k0 + 32 + asub);  a[4] = pa[0]; a[5] = pa[1]; a[6] = pa[2]; a[7] = pa[3];
#pragma unroll
    for (int jt = 0; jt < 4; ++jt) {
      const int* pb = &skp[((jt * 16 + r) * KPK_ + k0 + bsub) >> 2];
      v8i b;
      b[0] = pb[0]; b[1] = pb[1]; b[2] = pb[2];  b[3] = pb[3];
      b[4] = pb[8]; b[5] = pb[9]; b[6] = pb[10]; b[7] = pb[11];
      acc[jt] = __builtin_amdgcn_wmma_i32_16x16x64_iu8(false, a, false, b,
                                                       acc[jt], false, false);
    }
  }
  const float scale = 0.125f;
  float* out = scores + (size_t)(bh * S_ + ib * 64 + wave * 16) * S_;
#pragma unroll
  for (int jt = 0; jt < 4; ++jt) {
    int j = jb * 64 + jt * 16 + r;
#pragma unroll
    for (int v = 0; v < 8; ++v) {
      int ir = v + 8 * hi;
      out[(size_t)ir * S_ + j] = (float)acc[jt][v] * scale;
    }
  }
}

__global__ void asa_softmax(const float* __restrict__ scores,
                            _Float16* __restrict__ attn) {
  __shared__ float red[128];
  size_t row = blockIdx.x;
  const float* p = scores + row * S_;
  _Float16* po    = attn  + row * S_;
  int t = threadIdx.x;
  float vals[4];
  float m = -1e30f;
#pragma unroll
  for (int u = 0; u < 4; ++u) { vals[u] = p[t + u * 128]; m = fmaxf(m, vals[u]); }
  red[t] = m; __syncthreads();
  for (int s = 64; s > 0; s >>= 1) {
    if (t < s) red[t] = fmaxf(red[t], red[t + s]);
    __syncthreads();
  }
  m = red[0]; __syncthreads();
  float sum = 0.f;
#pragma unroll
  for (int u = 0; u < 4; ++u) { vals[u] = __expf(vals[u] - m); sum += vals[u]; }
  red[t] = sum; __syncthreads();
  for (int s = 64; s > 0; s >>= 1) {
    if (t < s) red[t] += red[t + s];
    __syncthreads();
  }
  float inv = 1.0f / red[0];
#pragma unroll
  for (int u = 0; u < 4; ++u) po[t + u * 128] = (_Float16)(vals[u] * inv);
}

__global__ void asa_av(const _Float16* __restrict__ attn,
                       const _Float16* __restrict__ vmT,
                       _Float16* __restrict__ out_head) {
  int tile = blockIdx.x;
  int dt = tile & 1;  tile >>= 1;
  int it = tile & 15; tile >>= 4;
  int bh = tile;
  int b = bh >> 3, h = bh & 7;
  int lane = threadIdx.x & 31;
  int r    = lane & 15;
  int hi   = lane >> 4;
  int abase = hi * 8;
  int bbase = hi * 16;

  const _Float16* ar0 = attn + (size_t)(bh * S_ + it * 32 + r) * S_;
  const _Float16* ar1 = ar0 + (size_t)16 * S_;
  const _Float16* br0 = vmT + (size_t)(bh * HD_ + dt * 32 + r) * S_;
  const _Float16* br1 = br0 + (size_t)16 * S_;

  v8f acc[2][2] = {};
  for (int k0 = 0; k0 < S_; k0 += 32) {
    v16h a0 = asa_afrag(ar0, k0, abase);
    v16h a1 = asa_afrag(ar1, k0, abase);
    v16h b0 = *(const v16h*)(br0 + k0 + bbase);
    v16h b1 = *(const v16h*)(br1 + k0 + bbase);
    acc[0][0] = __builtin_amdgcn_wmma_f32_16x16x32_f16(false, a0, false, b0,
                                                       (short)0, acc[0][0], false, false);
    acc[0][1] = __builtin_amdgcn_wmma_f32_16x16x32_f16(false, a0, false, b1,
                                                       (short)0, acc[0][1], false, false);
    acc[1][0] = __builtin_amdgcn_wmma_f32_16x16x32_f16(false, a1, false, b0,
                                                       (short)0, acc[1][0], false, false);
    acc[1][1] = __builtin_amdgcn_wmma_f32_16x16x32_f16(false, a1, false, b1,
                                                       (short)0, acc[1][1], false, false);
  }
#pragma unroll
  for (int ti = 0; ti < 2; ++ti) {
#pragma unroll
    for (int tj = 0; tj < 2; ++tj) {
#pragma unroll
      for (int v = 0; v < 8; ++v) {
        int ir = it * 32 + ti * 16 + v + 8 * hi;
        int d  = dt * 32 + tj * 16 + r;
        out_head[(size_t)(b * S_ + ir) * E_ + h * HD_ + d] =
            (_Float16)acc[ti][tj][v];
      }
    }
  }
}

extern "C" void kernel_launch(void* const* d_in, const int* in_sizes, int n_in,
                              void* d_out, int out_size, void* d_ws, size_t ws_size,
                              hipStream_t stream) {
  const float* x   = (const float*)d_in[0];
  const float* Wq  = (const float*)d_in[1];
  const float* Wk  = (const float*)d_in[2];
  const float* Wv  = (const float*)d_in[3];
  const float* Wo  = (const float*)d_in[4];
  const float* bo  = (const float*)d_in[5];
  const float* g1  = (const float*)d_in[6];
  const float* gb1 = (const float*)d_in[7];
  const float* g2  = (const float*)d_in[8];
  const float* gb2 = (const float*)d_in[9];
  const float* g3  = (const float*)d_in[10];
  const float* gb3 = (const float*)d_in[11];
  const float* c1  = (const float*)d_in[12];
  const float* cb1 = (const float*)d_in[13];
  const float* c2  = (const float*)d_in[14];
  const float* cb2 = (const float*)d_in[15];
  const float* alp = (const float*)d_in[16];
  const float* bet = (const float*)d_in[17];

  char* ws = (char*)d_ws;
  size_t off = 0;
  auto carve = [&](size_t bytes) -> char* {
    char* p = ws + off;
    off = (off + bytes + 255) & ~(size_t)255;
    return p;
  };
  _Float16* xh  = (_Float16*)carve((size_t)NTOK_ * E_ * 2);
  _Float16* WqT = (_Float16*)carve((size_t)E_ * E_ * 2);
  _Float16* WkT = (_Float16*)carve((size_t)E_ * E_ * 2);
  _Float16* WvT = (_Float16*)carve((size_t)E_ * E_ * 2);
  _Float16* WoT = (_Float16*)carve((size_t)E_ * E_ * 2);
  float* q      = (float*)carve((size_t)NTOK_ * E_ * 4);
  float* k      = (float*)carve((size_t)NTOK_ * E_ * 4);
  float* v      = (float*)carve((size_t)NTOK_ * E_ * 4);
  int*   Ti     = (int*)carve((size_t)NTOK_ * 4);
  signed char* qp = (signed char*)carve((size_t)B_ * H_ * S_ * KPK_);
  signed char* kp = (signed char*)carve((size_t)B_ * H_ * S_ * KPK_);
  _Float16* vmT   = (_Float16*)carve((size_t)B_ * H_ * HD_ * S_ * 2);
  float*    scores = (float*)carve((size_t)B_ * H_ * S_ * S_ * 4);
  _Float16* attnh  = (_Float16*)carve((size_t)B_ * H_ * S_ * S_ * 2);
  _Float16* outh   = (_Float16*)carve((size_t)NTOK_ * E_ * 2);

  asa_cvt_f16<<<(NTOK_ * E_ + 255) / 256, 256, 0, stream>>>(x, xh, NTOK_ * E_);
  int wgrid = (E_ * E_ + 255) / 256;
  asa_transpose_w_f16<<<wgrid, 256, 0, stream>>>(Wq, WqT);
  asa_transpose_w_f16<<<wgrid, 256, 0, stream>>>(Wk, WkT);
  asa_transpose_w_f16<<<wgrid, 256, 0, stream>>>(Wv, WvT);
  asa_transpose_w_f16<<<wgrid, 256, 0, stream>>>(Wo, WoT);

  asa_gate<<<NTOK_, 64, 0, stream>>>(x, g1, gb1, g2, gb2, g3, gb3,
                                     c1, cb1, c2, cb2, Ti);

  int ggrid = (NTOK_ / 32) * (E_ / 32);
  asa_gemm_f16<<<ggrid, 32, 0, stream>>>(xh, WqT, q, nullptr, NTOK_, E_, E_);
  asa_gemm_f16<<<ggrid, 32, 0, stream>>>(xh, WkT, k, nullptr, NTOK_, E_, E_);
  asa_gemm_f16<<<ggrid, 32, 0, stream>>>(xh, WvT, v, nullptr, NTOK_, E_, E_);

  asa_lif_pack<<<(NTOK_ * E_) / 256, 256, 0, stream>>>(q, k, v, Ti, qp, kp,
                                                       vmT, alp, bet);

  asa_scores_iu8<<<B_ * H_ * 8 * 8, 128, 0, stream>>>(qp, kp, scores);

  asa_softmax<<<B_ * H_ * S_, 128, 0, stream>>>(scores, attnh);

  asa_av<<<B_ * H_ * 16 * 2, 32, 0, stream>>>(attnh, vmT, outh);

  asa_gemm_f16<<<ggrid, 32, 0, stream>>>(outh, WoT, (float*)d_out, bo,
                                         NTOK_, E_, E_);
}
